// AttentionLayer_37787122270555
// MI455X (gfx1250) — matrix operand order repaired and checked by reading, not compiled, not run on hardware
//
#include <hip/hip_runtime.h>

typedef __attribute__((ext_vector_type(16))) __bf16 v16bf;
typedef __attribute__((ext_vector_type(8)))  float  v8f;
typedef __attribute__((ext_vector_type(2)))  float  v2f;

#define N_NODES 8192
#define D_IN    256
#define F_OUT   64
#define LEAKY   0.2f

// Order-preserving float<->int mapping so we can use integer atomicMax for a
// global float max (s2_max).
__device__ __forceinline__ int f32_to_ord(float f) {
    int i = __float_as_int(f);
    return i >= 0 ? i : (int)(0x80000000u - (unsigned)i);
}
__device__ __forceinline__ float ord_to_f32(int o) {
    int i = o >= 0 ? o : (int)(0x80000000u - (unsigned)o);
    return __int_as_float(i);
}

__global__ void gat_init(int* s2max_ord) {
    *s2max_ord = (int)0x80000000;   // floor of the ordered-int domain
}

// ---------------------------------------------------------------------------
// Kernel A: h = X @ W in full fp32 via V_WMMA_F32_16X16X4_F32.
// One wave per 16x16 C tile; 4 waves/block cover F=64.
// A layout (32-bit 16x4): lanes 0-15 hold K=kb+0,kb+1; lanes 16-31 K=kb+2,kb+3.
// ---------------------------------------------------------------------------
__global__ __launch_bounds__(128) void gat_h_gemm(const float* __restrict__ X,
                                                  const float* __restrict__ W,
                                                  float* __restrict__ H) {
    const int  lane = threadIdx.x & 31;
    const int  wv   = threadIdx.x >> 5;        // feature tile 0..3
    const bool lo   = lane < 16;
    const int  m    = lane & 15;
    const int  rowBase = blockIdx.x * 16;
    const int  fbase   = wv * 16;
    const float* xrow  = X + (size_t)(rowBase + m) * D_IN;

    v8f c = {0.f, 0.f, 0.f, 0.f, 0.f, 0.f, 0.f, 0.f};
    #pragma unroll 4
    for (int kb = 0; kb < D_IN; kb += 4) {
        const int ksub = kb + (lo ? 0 : 2);
        v2f a = *(const v2f*)(xrow + ksub);              // contiguous 8B load
        v2f b;
        b.x = W[(size_t)ksub * F_OUT + fbase + m];       // W resident in cache
        b.y = W[(size_t)(ksub + 1) * F_OUT + fbase + m];
        c = __builtin_amdgcn_wmma_f32_16x16x4_f32(false, a, false, b,
                                                  (short)0, c, false, false);
    }
    // C layout: VGPR r -> row r (lanes 0-15) / row r+8 (lanes 16-31), col = m
    #pragma unroll
    for (int r = 0; r < 8; ++r) {
        const int row = rowBase + r + (lo ? 0 : 8);
        H[(size_t)row * F_OUT + fbase + m] = c[r];
    }
}

// ---------------------------------------------------------------------------
// Kernel A2: s1 = h@a1, s2 = h@a2, global max(s2), and hT = bf16 transpose of h
// (so the attention GEMM's B operand loads are lane-contiguous).
// ---------------------------------------------------------------------------
__global__ __launch_bounds__(256) void gat_post(const float* __restrict__ H,
                                                const float* __restrict__ avec,
                                                float* __restrict__ s1,
                                                float* __restrict__ s2,
                                                int* __restrict__ s2max_ord,
                                                __bf16* __restrict__ hT) {
    const int i = blockIdx.x * blockDim.x + threadIdx.x;
    float acc1 = 0.f, acc2 = 0.f;
    #pragma unroll 8
    for (int f = 0; f < F_OUT; ++f) {
        const float hv = H[(size_t)i * F_OUT + f];
        acc1 = fmaf(hv, avec[f], acc1);
        acc2 = fmaf(hv, avec[F_OUT + f], acc2);
        hT[(size_t)f * N_NODES + i] = (__bf16)hv;
    }
    s1[i] = acc1;
    s2[i] = acc2;
    atomicMax(s2max_ord, f32_to_ord(acc2));
}

// ---------------------------------------------------------------------------
// Kernel B: fused mask + softmax + attn@h + ELU.
// Shift trick: m_i = leaky_relu(s1[i] + max_j s2[j]) >= every score in row i,
// so exp(e - m_i) never overflows and no online rescaling is needed.
// 512 blocks x 8 waves; wave handles j tiles of 32, 4x v_wmma_f32_16x16x32_bf16
// per tile; LDS reduction across the 8 waves at the end.
// ---------------------------------------------------------------------------
__global__ __launch_bounds__(256) void gat_attn(const int* __restrict__ adj,
                                                const float* __restrict__ s1,
                                                const float* __restrict__ s2,
                                                const int* __restrict__ s2max_ord,
                                                const __bf16* __restrict__ hT,
                                                float* __restrict__ out) {
    __shared__ float red[8][16][F_OUT];   // 32 KB  per-wave partial C tiles
    __shared__ float redsum[8][32];       // 1 KB   per-lane weight sums

    const int  wave = threadIdx.x >> 5;
    const int  lane = threadIdx.x & 31;
    const bool lo   = lane < 16;
    const int  m    = lane & 15;
    const int  row  = blockIdx.x * 16 + m;

    const float s1i   = s1[row];
    const float s2max = ord_to_f32(*s2max_ord);
    float mi = s1i + s2max;
    mi = mi > 0.f ? mi : LEAKY * mi;

    const int* adjrow = adj + (size_t)row * N_NODES;

    v8f c0 = {0.f,0.f,0.f,0.f,0.f,0.f,0.f,0.f};
    v8f c1 = c0, c2 = c0, c3 = c0;
    float wsum = 0.f;

    for (int jb = wave * 32; jb < N_NODES; jb += 256) {
        // A-operand lane order (16-bit 16x32): lanes 0-15 own K = jb+{0..15},
        // lanes 16-31 own K = jb+{16..31}.
        const int base = jb + (lo ? 0 : 16);

        const int4 q0 = *(const int4*)(adjrow + base);
        const int4 q1 = *(const int4*)(adjrow + base + 4);
        const int4 q2 = *(const int4*)(adjrow + base + 8);
        const int4 q3 = *(const int4*)(adjrow + base + 12);
        if (jb + 256 < N_NODES)
            __builtin_prefetch(adjrow + base + 256, 0, 1);  // global_prefetch_b8

        const float4 p0 = *(const float4*)(s2 + base);
        const float4 p1 = *(const float4*)(s2 + base + 4);
        const float4 p2 = *(const float4*)(s2 + base + 8);
        const float4 p3 = *(const float4*)(s2 + base + 12);

        const int   av[16] = {q0.x,q0.y,q0.z,q0.w, q1.x,q1.y,q1.z,q1.w,
                              q2.x,q2.y,q2.z,q2.w, q3.x,q3.y,q3.z,q3.w};
        const float sv[16] = {p0.x,p0.y,p0.z,p0.w, p1.x,p1.y,p1.z,p1.w,
                              p2.x,p2.y,p2.z,p2.w, p3.x,p3.y,p3.z,p3.w};

        v16bf aw;
        #pragma unroll
        for (int t = 0; t < 16; ++t) {
            float e = s1i + sv[t];
            e = e > 0.f ? e : LEAKY * e;
            const float w = (av[t] > 0) ? __expf(e - mi) : 0.f;
            wsum += w;
            aw[t] = (__bf16)w;
        }

        // B-operand (16-bit 32x16): lane = column m, lanes 0-15 hold K=jb+0..15,
        // lanes 16-31 hold K=jb+16..31 -> 16 contiguous bf16 from hT per lane.
        const int koff = jb + (lo ? 0 : 16);
        const v16bf b0 = *(const v16bf*)(hT + (size_t)(m)      * N_NODES + koff);
        const v16bf b1 = *(const v16bf*)(hT + (size_t)(16 + m) * N_NODES + koff);
        const v16bf b2 = *(const v16bf*)(hT + (size_t)(32 + m) * N_NODES + koff);
        const v16bf b3 = *(const v16bf*)(hT + (size_t)(48 + m) * N_NODES + koff);

        c0 = __builtin_amdgcn_wmma_f32_16x16x32_bf16(false, aw, false, b0, (short)0, c0, false, false);
        c1 = __builtin_amdgcn_wmma_f32_16x16x32_bf16(false, aw, false, b1, (short)0, c1, false, false);
        c2 = __builtin_amdgcn_wmma_f32_16x16x32_bf16(false, aw, false, b2, (short)0, c2, false, false);
        c3 = __builtin_amdgcn_wmma_f32_16x16x32_bf16(false, aw, false, b3, (short)0, c3, false, false);
    }

    redsum[wave][lane] = wsum;
    #pragma unroll
    for (int r = 0; r < 8; ++r) {
        const int rr = r + (lo ? 0 : 8);
        red[wave][rr][ 0 + m] = c0[r];
        red[wave][rr][16 + m] = c1[r];
        red[wave][rr][32 + m] = c2[r];
        red[wave][rr][48 + m] = c3[r];
    }
    __syncthreads();

    // Reduce 8 waves -> normalize -> ELU -> store. 1024 outputs / 256 threads.
    #pragma unroll
    for (int k = 0; k < 4; ++k) {
        const int elem = threadIdx.x + k * 256;
        const int r    = elem >> 6;
        const int col  = elem & 63;
        float acc = 0.f, rs = 0.f;
        #pragma unroll
        for (int w = 0; w < 8; ++w) {
            acc += red[w][r][col];
            rs  += redsum[w][r] + redsum[w][r + 16];
        }
        float val = acc / rs;
        val = val > 0.f ? val : (__expf(val) - 1.f);
        out[(size_t)(blockIdx.x * 16 + r) * F_OUT + col] = val;
    }
}

extern "C" void kernel_launch(void* const* d_in, const int* in_sizes, int n_in,
                              void* d_out, int out_size, void* d_ws, size_t ws_size,
                              hipStream_t stream) {
    const float* features = (const float*)d_in[0];   // 8192 x 256 f32
    const int*   adj      = (const int*)d_in[1];     // 8192 x 8192 i32
    const float* W        = (const float*)d_in[2];   // 256 x 64 f32
    const float* avec     = (const float*)d_in[3];   // 128 f32
    float* out = (float*)d_out;                      // 8192 x 64 f32

    char* ws = (char*)d_ws;
    float*  H     = (float*)ws;                                   // 2 MB
    __bf16* hT    = (__bf16*)(ws + (size_t)2 * 1024 * 1024);      // 1 MB
    float*  s1    = (float*)(ws + (size_t)3 * 1024 * 1024);       // 32 KB
    float*  s2    = (float*)(ws + (size_t)3 * 1024 * 1024 + 32 * 1024);
    int*    s2max = (int*)  (ws + (size_t)3 * 1024 * 1024 + 64 * 1024);

    gat_init  <<<1, 1, 0, stream>>>(s2max);
    gat_h_gemm<<<N_NODES / 16, 128, 0, stream>>>(features, W, H);
    gat_post  <<<N_NODES / 256, 256, 0, stream>>>(H, avec, s1, s2, s2max, hT);
    gat_attn  <<<N_NODES / 16, 256, 0, stream>>>(adj, s1, s2, s2max, hT, out);
}
